// CrossAttention_84318797955433
// MI455X (gfx1250) — matrix operand order repaired and checked by reading, not compiled, not run on hardware
//
#include <hip/hip_runtime.h>
#include <stdint.h>

typedef __attribute__((ext_vector_type(16))) _Float16 v16h;
typedef __attribute__((ext_vector_type(8)))  float    v8f;
typedef _Float16 hf;

#define Bz  4
#define Lz  2048
#define Hz  1024
#define NHz 16
#define DHz 64

union Frag { v16h v; uint32_t u[8]; };

__device__ __forceinline__ v8f wmma16(const Frag& a, const Frag& b, v8f c) {
  return __builtin_amdgcn_wmma_f32_16x16x32_f16(false, a.v, false, b.v,
                                                (short)0, c, false, false);
}

#if __has_builtin(__builtin_amdgcn_s_wait_asynccnt)
#define WAIT_ASYNC0() __builtin_amdgcn_s_wait_asynccnt(0)
#else
#define WAIT_ASYNC0() asm volatile("s_wait_asynccnt 0" ::: "memory")
#endif

// Async DMA of 16 bytes global -> LDS (CDNA5 GLOBAL_LOAD_ASYNC_TO_LDS_B128,
// tracked on ASYNCcnt; no VGPR round trip).
__device__ __forceinline__ void async_copy16(const void* gsrc, void* ldst) {
  uint32_t lds = (uint32_t)(uintptr_t)ldst;        // low 32 bits = LDS offset
  uint64_t ga  = (uint64_t)(uintptr_t)gsrc;
  asm volatile("global_load_async_to_lds_b128 %0, %1, off"
               :: "v"(lds), "v"(ga) : "memory");
}

// ---------------------------------------------------------------------------
// fp32 -> f16 bulk convert (used once for the four weight matrices).
// ---------------------------------------------------------------------------
__global__ __launch_bounds__(256)
void cvt_f16_kernel(const float* __restrict__ src, hf* __restrict__ dst, int n)
{
  const int i = (blockIdx.x * 256 + threadIdx.x) * 8;
  if (i + 8 <= n) {
    const float4* p = (const float4*)(src + i);
    float4 a = p[0], b = p[1];
    dst[i+0]=(hf)a.x; dst[i+1]=(hf)a.y; dst[i+2]=(hf)a.z; dst[i+3]=(hf)a.w;
    dst[i+4]=(hf)b.x; dst[i+5]=(hf)b.y; dst[i+6]=(hf)b.z; dst[i+7]=(hf)b.w;
  }
}

// ---------------------------------------------------------------------------
// GEMM  Y = X(fp32) @ Wh(f16)^T + bias, emitted f16 in head-major layouts.
// mode 0/1: Y[b][head][l][dh]   (Q,K)      mode 2: Y[b][head][dh][l]  (V^T)
// 256 thr = 8 waves, 64x64 tile, K staged 32/step, double-buffered LDS;
// W tile copied by async-to-LDS DMA, X tile converted f32->f16 in staging.
// ---------------------------------------------------------------------------
__global__ __launch_bounds__(256)
void proj_wmma_kernel(const float* __restrict__ X, const hf* __restrict__ Wh,
                      const float* __restrict__ bias, hf* __restrict__ Y, int mode)
{
  __shared__ hf As[2][64][40];
  __shared__ hf Bs[2][64][40];
  const int tid  = threadIdx.x;
  const int lane = tid & 31, wave = tid >> 5;
  const int lrow = lane & 15, hi = lane >> 4;
  const int nbase = blockIdx.x * 64;
  const int mbase = blockIdx.y * 64;
  const int mi = wave >> 1;            // M subtile 0..3
  const int nA = (wave & 1) * 32;      // first of two N subtiles
  const int sr = tid >> 2;             // staging row 0..63
  const int sc = (tid & 3) * 8;        // staging col 0/8/16/24
  const int klo = hi * 16;             // A-fragment K grouping
  const int kb2 = hi * 16;             // B-fragment K grouping

  v8f acc[2];
#pragma unroll
  for (int s = 0; s < 2; ++s)
#pragma unroll
    for (int e = 0; e < 8; ++e) acc[s][e] = 0.0f;

  auto stage = [&](int kt, int buf) {
    const int k0 = kt * 32;
    async_copy16(Wh + (size_t)(nbase + sr) * Hz + k0 + sc, &Bs[buf][sr][sc]);
    const float4* px = (const float4*)(X + (size_t)(mbase + sr) * Hz + k0 + sc);
    float4 x0 = px[0], x1 = px[1];
    hf* d = &As[buf][sr][sc];
    d[0]=(hf)x0.x; d[1]=(hf)x0.y; d[2]=(hf)x0.z; d[3]=(hf)x0.w;
    d[4]=(hf)x1.x; d[5]=(hf)x1.y; d[6]=(hf)x1.z; d[7]=(hf)x1.w;
  };

  stage(0, 0);
  WAIT_ASYNC0();
  __syncthreads();

  const int KT = Hz / 32;
  for (int kt = 0; kt < KT; ++kt) {
    const int cur = kt & 1;
    if (kt + 1 < KT) stage(kt + 1, cur ^ 1);

    Frag a;
#pragma unroll
    for (int i = 0; i < 4; ++i) {
      a.u[i]     = *(const uint32_t*)&As[cur][mi*16 + lrow][klo + 2*i];
      a.u[4 + i] = *(const uint32_t*)&As[cur][mi*16 + lrow][8 + klo + 2*i];
    }
    Frag b0, b1;
#pragma unroll
    for (int i = 0; i < 8; ++i) {
      b0.u[i] = *(const uint32_t*)&Bs[cur][nA      + lrow][kb2 + 2*i];
      b1.u[i] = *(const uint32_t*)&Bs[cur][nA + 16 + lrow][kb2 + 2*i];
    }
    acc[0] = wmma16(a, b0, acc[0]);
    acc[1] = wmma16(a, b1, acc[1]);

    if (kt + 1 < KT) WAIT_ASYNC0();
    __syncthreads();
  }

#pragma unroll
  for (int sub = 0; sub < 2; ++sub) {
    const int n = nbase + nA + sub*16 + lrow;
    const float bval = bias[n];
    const int head = n >> 6, dc = n & 63;
#pragma unroll
    for (int j = 0; j < 8; ++j) {
      const int m  = mbase + mi*16 + j + 8*hi;
      const int bb = m >> 11, ll = m & (Lz - 1);
      const float val = acc[sub][j] + bval;
      size_t idx;
      if (mode == 2) idx = (((size_t)(bb*NHz + head))*DHz + dc)*Lz + ll;
      else           idx = (((size_t)(bb*NHz + head))*Lz + ll)*DHz + dc;
      Y[idx] = (hf)val;
    }
  }
}

// ---------------------------------------------------------------------------
// Flash attention: each wave owns 16 q-rows, streams 32 keys per step.
// S = Q K^T / 8 (4 WMMAs), online softmax (shfl-xor row reductions in the
// D-layout lane halves), P redistributed D->A layout via wave-private LDS,
// O += P V (4 WMMAs). Query-row masking per the reference semantics.
// ---------------------------------------------------------------------------
__global__ __launch_bounds__(256)
void attn_wmma_kernel(const hf* __restrict__ Qh, const hf* __restrict__ Kh,
                      const hf* __restrict__ VhT, const int* __restrict__ mask,
                      hf* __restrict__ Hid)
{
  __shared__ hf Pbuf[8][16][34];
  const int tid  = threadIdx.x;
  const int lane = tid & 31, wave = tid >> 5;
  const int lrow = lane & 15, hi = lane >> 4;
  const int b = blockIdx.z, h = blockIdx.y;
  const int qbase = blockIdx.x * 128 + wave * 16;
  const int klo = hi * 16;

  const hf* Qp = Qh  + (((size_t)(b*NHz + h))*Lz + qbase) * DHz;
  const hf* Kp = Kh  + ((size_t)(b*NHz + h)) * Lz * DHz;
  const hf* Vp = VhT + ((size_t)(b*NHz + h)) * DHz * Lz;

  Frag aq[2];                                   // Q, dh chunks [0,32) [32,64)
#pragma unroll
  for (int c = 0; c < 2; ++c)
#pragma unroll
    for (int i = 0; i < 4; ++i) {
      aq[c].u[i]     = *(const uint32_t*)(Qp + lrow*DHz + c*32 + klo + 2*i);
      aq[c].u[4 + i] = *(const uint32_t*)(Qp + lrow*DHz + c*32 + 8 + klo + 2*i);
    }

  bool qok[8];
#pragma unroll
  for (int j = 0; j < 8; ++j)
    qok[j] = mask[b*Lz + qbase + j + 8*hi] != 0;

  float rm[8], rl[8];
  v8f o[4];
#pragma unroll
  for (int j = 0; j < 8; ++j) { rm[j] = -3.0e38f; rl[j] = 0.0f; }
#pragma unroll
  for (int nt = 0; nt < 4; ++nt)
#pragma unroll
    for (int e = 0; e < 8; ++e) o[nt][e] = 0.0f;

  for (int kt = 0; kt < Lz / 32; ++kt) {
    const int kb = kt * 32;
    if (kt + 1 < Lz / 32)
      __builtin_prefetch(Kp + (size_t)(kb + 32) * DHz, 0, 1);

    v8f s[2];
#pragma unroll
    for (int sub = 0; sub < 2; ++sub) {
      Frag bk0, bk1;
      const int key = kb + sub*16 + lrow;
      const int dhb = hi * 16;
#pragma unroll
      for (int i = 0; i < 8; ++i) {
        bk0.u[i] = *(const uint32_t*)(Kp + (size_t)key*DHz + dhb + 2*i);
        bk1.u[i] = *(const uint32_t*)(Kp + (size_t)key*DHz + 32 + dhb + 2*i);
      }
      v8f z;
#pragma unroll
      for (int e = 0; e < 8; ++e) z[e] = 0.0f;
      z = wmma16(aq[0], bk0, z);
      z = wmma16(aq[1], bk1, z);
      s[sub] = z;
    }

    float p0[8], p1[8];
#pragma unroll
    for (int j = 0; j < 8; ++j) {
      float f0 = qok[j] ? s[0][j] * 0.125f : -1.0e9f;
      float f1 = qok[j] ? s[1][j] * 0.125f : -1.0e9f;
      float tm = fmaxf(f0, f1);
#pragma unroll
      for (int off = 1; off < 16; off <<= 1)
        tm = fmaxf(tm, __shfl_xor(tm, off, 32));
      const float newm = fmaxf(rm[j], tm);
      const float al = __expf(rm[j] - newm);
      p0[j] = __expf(f0 - newm);
      p1[j] = __expf(f1 - newm);
      float rs = p0[j] + p1[j];
#pragma unroll
      for (int off = 1; off < 16; off <<= 1)
        rs += __shfl_xor(rs, off, 32);
      rl[j] = rl[j] * al + rs;
      rm[j] = newm;
#pragma unroll
      for (int nt = 0; nt < 4; ++nt) o[nt][j] *= al;
      Pbuf[wave][j + 8*hi][lrow]      = (hf)p0[j];
      Pbuf[wave][j + 8*hi][16 + lrow] = (hf)p1[j];
    }
    asm volatile("s_wait_dscnt 0" ::: "memory");

    Frag pa;                                    // P as A operand (16x32 keys)
#pragma unroll
    for (int i = 0; i < 4; ++i) {
      pa.u[i]     = *(const uint32_t*)&Pbuf[wave][lrow][klo + 2*i];
      pa.u[4 + i] = *(const uint32_t*)&Pbuf[wave][lrow][8 + klo + 2*i];
    }
    const int keyb = kb + hi * 16;
#pragma unroll
    for (int nt = 0; nt < 4; ++nt) {
      Frag bvf;
      const int dhcol = nt*16 + lrow;
#pragma unroll
      for (int i = 0; i < 8; ++i)
        bvf.u[i] = *(const uint32_t*)(Vp + (size_t)dhcol*Lz + keyb + 2*i);
      o[nt] = wmma16(pa, bvf, o[nt]);
    }
  }

  float inv[8];
#pragma unroll
  for (int j = 0; j < 8; ++j) inv[j] = 1.0f / rl[j];
#pragma unroll
  for (int nt = 0; nt < 4; ++nt)
#pragma unroll
    for (int j = 0; j < 8; ++j) {
      const size_t row = (size_t)b * Lz + qbase + j + 8*hi;
      const int col = h*DHz + nt*16 + lrow;
      Hid[row * Hz + col] = (hf)(o[nt][j] * inv[j]);
    }
}

// ---------------------------------------------------------------------------
// Out = res(fp32) + Hid(f16) @ Woh(f16)^T + bo, fp32 output.
// Both tiles are f16 in memory -> both staged by async-to-LDS DMA,
// double-buffered; zero staging VALU.
// ---------------------------------------------------------------------------
__global__ __launch_bounds__(256)
void outproj_wmma_kernel(const hf* __restrict__ Hid, const hf* __restrict__ Wh,
                         const float* __restrict__ bias, const float* __restrict__ res,
                         float* __restrict__ Out)
{
  __shared__ hf As[2][64][40];
  __shared__ hf Bs[2][64][40];
  const int tid  = threadIdx.x;
  const int lane = tid & 31, wave = tid >> 5;
  const int lrow = lane & 15, hi = lane >> 4;
  const int nbase = blockIdx.x * 64;
  const int mbase = blockIdx.y * 64;
  const int mi = wave >> 1;
  const int nA = (wave & 1) * 32;
  const int sr = tid >> 2;
  const int sc = (tid & 3) * 8;
  const int klo = hi * 16;
  const int kb2 = hi * 16;

  v8f acc[2];
#pragma unroll
  for (int s = 0; s < 2; ++s)
#pragma unroll
    for (int e = 0; e < 8; ++e) acc[s][e] = 0.0f;

  auto stage = [&](int kt, int buf) {
    const int k0 = kt * 32;
    async_copy16(Hid + (size_t)(mbase + sr) * Hz + k0 + sc, &As[buf][sr][sc]);
    async_copy16(Wh  + (size_t)(nbase + sr) * Hz + k0 + sc, &Bs[buf][sr][sc]);
  };

  stage(0, 0);
  WAIT_ASYNC0();
  __syncthreads();

  const int KT = Hz / 32;
  for (int kt = 0; kt < KT; ++kt) {
    const int cur = kt & 1;
    if (kt + 1 < KT) stage(kt + 1, cur ^ 1);

    Frag a;
#pragma unroll
    for (int i = 0; i < 4; ++i) {
      a.u[i]     = *(const uint32_t*)&As[cur][mi*16 + lrow][klo + 2*i];
      a.u[4 + i] = *(const uint32_t*)&As[cur][mi*16 + lrow][8 + klo + 2*i];
    }
    Frag b0, b1;
#pragma unroll
    for (int i = 0; i < 8; ++i) {
      b0.u[i] = *(const uint32_t*)&Bs[cur][nA      + lrow][kb2 + 2*i];
      b1.u[i] = *(const uint32_t*)&Bs[cur][nA + 16 + lrow][kb2 + 2*i];
    }
    acc[0] = wmma16(a, b0, acc[0]);
    acc[1] = wmma16(a, b1, acc[1]);

    if (kt + 1 < KT) WAIT_ASYNC0();
    __syncthreads();
  }

#pragma unroll
  for (int sub = 0; sub < 2; ++sub) {
    const int n = nbase + nA + sub*16 + lrow;
    const float bval = bias[n];
#pragma unroll
    for (int j = 0; j < 8; ++j) {
      const int m = mbase + mi*16 + j + 8*hi;
      const size_t idx = (size_t)m * Hz + n;
      Out[idx] = res[idx] + acc[sub][j] + bval;
    }
  }
}

// ---------------------------------------------------------------------------
extern "C" void kernel_launch(void* const* d_in, const int* in_sizes, int n_in,
                              void* d_out, int out_size, void* d_ws, size_t ws_size,
                              hipStream_t stream) {
  (void)in_sizes; (void)n_in; (void)out_size; (void)ws_size;
  const float* q  = (const float*)d_in[0];
  const float* k  = (const float*)d_in[1];
  const float* v  = (const float*)d_in[2];
  const int* mask = (const int*)d_in[3];
  const float* wq = (const float*)d_in[4];
  const float* bq = (const float*)d_in[5];
  const float* wk = (const float*)d_in[6];
  const float* bk = (const float*)d_in[7];
  const float* wv = (const float*)d_in[8];
  const float* bv = (const float*)d_in[9];
  const float* wo = (const float*)d_in[10];
  const float* bo = (const float*)d_in[11];

  char* ws = (char*)d_ws;
  const size_t SQ = (size_t)Bz * NHz * Lz * DHz * sizeof(hf);   // 16 MiB
  const size_t SW = (size_t)Hz * Hz * sizeof(hf);               //  2 MiB
  hf* Qh  = (hf*)(ws);
  hf* Kh  = (hf*)(ws + SQ);
  hf* VhT = (hf*)(ws + 2 * SQ);
  hf* Hid = (hf*)(ws + 3 * SQ);
  hf* Wqh = (hf*)(ws + 4 * SQ);
  hf* Wkh = (hf*)(ws + 4 * SQ + SW);
  hf* Wvh = (hf*)(ws + 4 * SQ + 2 * SW);
  hf* Woh = (hf*)(ws + 4 * SQ + 3 * SW);

  dim3 blk(256);
  const int nW = Hz * Hz;
  dim3 gcvt(nW / (256 * 8));
  cvt_f16_kernel<<<gcvt, blk, 0, stream>>>(wq, Wqh, nW);
  cvt_f16_kernel<<<gcvt, blk, 0, stream>>>(wk, Wkh, nW);
  cvt_f16_kernel<<<gcvt, blk, 0, stream>>>(wv, Wvh, nW);
  cvt_f16_kernel<<<gcvt, blk, 0, stream>>>(wo, Woh, nW);

  dim3 gproj(Hz / 64, (Bz * Lz) / 64);
  proj_wmma_kernel<<<gproj, blk, 0, stream>>>(q, Wqh, bq, Qh, 0);
  proj_wmma_kernel<<<gproj, blk, 0, stream>>>(k, Wkh, bk, Kh, 1);
  proj_wmma_kernel<<<gproj, blk, 0, stream>>>(v, Wvh, bv, VhT, 2);

  dim3 gattn(Lz / 128, NHz, Bz);
  attn_wmma_kernel<<<gattn, blk, 0, stream>>>(Qh, Kh, VhT, mask, Hid);

  outproj_wmma_kernel<<<gproj, blk, 0, stream>>>(Hid, Woh, bo, k, (float*)d_out);
}
